// AttentionBlock3D_21629455302875
// MI455X (gfx1250) — matrix operand order repaired and checked by reading, not compiled, not run on hardware
//
#include <hip/hip_runtime.h>
#include <hip/hip_bf16.h>
#include <math.h>

// ---------------------------------------------------------------------------
// AttentionBlock3D for gfx1250 (MI455X), wave32 + v_wmma_f32_16x16x32_f16.
// B=2, C=512, D=8, H=16, W=16 -> N=2048, HEADS=8, HEAD_DIM=64, GROUPS=8.
// ---------------------------------------------------------------------------

typedef __attribute__((ext_vector_type(16))) _Float16 v16h;
typedef __attribute__((ext_vector_type(8)))  float    v8f;

#define CB   2
#define CC   512
#define CN   2048            // D*H*W
#define CHEADS 8
#define CHD  64

__device__ __forceinline__ v8f wmma16(v16h a, v16h b, v8f c) {
  // D = A(16x32,f16) * B(32x16,f16) + C(16x16,f32)
  return __builtin_amdgcn_wmma_f32_16x16x32_f16(
      /*neg_a=*/false, a, /*neg_b=*/false, b,
      /*c_mod=*/(short)0, c, /*reuse_a=*/false, /*reuse_b=*/false);
}

// A-operand (16x32 f16) load, row-major source, row m = lane&15.
// Layout: lanes 0-15 hold K 0..15, lanes 16-31 hold K 16..31 (contiguous, matches load_b16).
__device__ __forceinline__ v16h load_a16(const _Float16* base, int ld, int lane) {
  const int m = lane & 15, hi = (lane >> 4) & 1;
  const _Float16* p = base + (size_t)m * ld;
  v16h r;
#pragma unroll
  for (int j = 0; j < 16; ++j) r[j] = p[hi * 16 + j];
  return r;
}

// B-operand (32x16 f16) load. Memory holds B^T row-major (column n contiguous
// in K with stride ld). Lane n holds K = hi*16 .. hi*16+15 contiguously.
__device__ __forceinline__ v16h load_b16(const _Float16* base, int ld, int lane) {
  const int n = lane & 15, hi = (lane >> 4) & 1;
  const _Float16* p = base + (size_t)n * ld + hi * 16;
  v16h r;
#pragma unroll
  for (int j = 0; j < 16; ++j) r[j] = p[j];
  return r;
}

__device__ __forceinline__ v8f zero8() {
  v8f z;
#pragma unroll
  for (int j = 0; j < 8; ++j) z[j] = 0.0f;
  return z;
}

// ---------------------------------------------------------------------------
// fp32 -> f16 convert (weights)
// ---------------------------------------------------------------------------
__global__ void cvt_f32_f16(const float* __restrict__ in, _Float16* __restrict__ out, int n) {
  int i = blockIdx.x * 256 + threadIdx.x;
  if (i < n) out[i] = (_Float16)in[i];
}

// ---------------------------------------------------------------------------
// GroupNorm(8, 512) -> write h transposed [b][n][c] as f16 for GEMM B-operand
// One block per (b, group): 64 channels * 2048 elems.
// ---------------------------------------------------------------------------
__global__ void gn_kernel(const float* __restrict__ x, const float* __restrict__ gw,
                          const float* __restrict__ gb, _Float16* __restrict__ ht) {
  const int b = blockIdx.x >> 3, g = blockIdx.x & 7;
  const float* xp = x + (((size_t)b * CC) + g * 64) * CN;
  float s = 0.f, s2 = 0.f;
  for (int i = threadIdx.x; i < 64 * CN; i += 256) {
    float v = xp[i]; s += v; s2 += v * v;
  }
  __shared__ float sh1[256], sh2[256];
  sh1[threadIdx.x] = s; sh2[threadIdx.x] = s2;
  __syncthreads();
  for (int off = 128; off > 0; off >>= 1) {
    if (threadIdx.x < off) {
      sh1[threadIdx.x] += sh1[threadIdx.x + off];
      sh2[threadIdx.x] += sh2[threadIdx.x + off];
    }
    __syncthreads();
  }
  const float inv_cnt = 1.0f / (64.0f * (float)CN);
  const float mean = sh1[0] * inv_cnt;
  const float var  = sh2[0] * inv_cnt - mean * mean;
  const float rstd = rsqrtf(var + 1e-5f);
  for (int i = threadIdx.x; i < 64 * CN; i += 256) {
    const int cl = i >> 11;            // i / 2048
    const int n  = i & (CN - 1);
    const int c  = g * 64 + cl;
    float v = (xp[i] - mean) * rstd * gw[c] + gb[c];
    ht[((size_t)b * CN + n) * CC + c] = (_Float16)v;
  }
}

// ---------------------------------------------------------------------------
// QKV GEMM: out[b][o][n] = sum_c W[o][c] * h[b][c][n] + bias[o]
// A = qkv_w f16 [1536x512] row-major, B = h^T f16 [b][n][512].
// Wave tile: 16(M) x 64(N). Scatter into Q,K [b][head][n][64] and V^T
// [b][head][64][n] (all f16) for the attention kernel's operand layouts.
// ---------------------------------------------------------------------------
__global__ void qkv_gemm(const _Float16* __restrict__ w16, const _Float16* __restrict__ ht,
                         const float* __restrict__ qkvb,
                         _Float16* __restrict__ qb, _Float16* __restrict__ kb,
                         _Float16* __restrict__ vt) {
  const int wave = blockIdx.x * 8 + (threadIdx.x >> 5);
  const int lane = threadIdx.x & 31;
  const int b = wave / 3072;            // 96 Mtiles * 32 Ngroups
  const int rem = wave % 3072;
  const int mt = rem >> 5, nt = rem & 31;
  const _Float16* hb = ht + (size_t)b * CN * CC;

  v8f acc[4];
#pragma unroll
  for (int nn = 0; nn < 4; ++nn) acc[nn] = zero8();

  for (int k0 = 0; k0 < CC; k0 += 32) {
    v16h a = load_a16(w16 + (size_t)(mt * 16) * CC + k0, CC, lane);
#pragma unroll
    for (int nn = 0; nn < 4; ++nn) {
      v16h bm = load_b16(hb + (size_t)(nt * 64 + nn * 16) * CC + k0, CC, lane);
      acc[nn] = wmma16(a, bm, acc[nn]);
    }
  }

  const int col = lane & 15, hi = lane >> 4;
#pragma unroll
  for (int nn = 0; nn < 4; ++nn) {
    const int n = nt * 64 + nn * 16 + col;
#pragma unroll
    for (int r = 0; r < 8; ++r) {
      const int o = mt * 16 + r + 8 * hi;
      const float v = acc[nn][r] + qkvb[o];
      const int which = o >> 9;          // 0:q 1:k 2:v
      const int oc = o & 511;
      const int head = oc >> 6, dd = oc & 63;
      const size_t hb2 = (size_t)b * CHEADS + head;
      if (which == 0)      qb[(hb2 * CN + n) * CHD + dd] = (_Float16)v;
      else if (which == 1) kb[(hb2 * CN + n) * CHD + dd] = (_Float16)v;
      else                 vt[(hb2 * CHD + dd) * CN + n] = (_Float16)v;
    }
  }
}

// ---------------------------------------------------------------------------
// Flash attention, one wave = 16 query rows, online softmax over 32-key tiles.
// Relative-position bias computed on the fly (bucket = floor(dist/4), table
// in LDS). P-tile transposed accumulator->A-operand via per-wave LDS.
// Output written transposed [b][n][c] f16 for the projection GEMM.
// ---------------------------------------------------------------------------
__global__ void attn_kernel(const _Float16* __restrict__ qb, const _Float16* __restrict__ kb,
                            const _Float16* __restrict__ vt, const float* __restrict__ rel,
                            _Float16* __restrict__ at) {
  __shared__ float   relS[32];
  __shared__ _Float16 pS[8][16 * 32];   // per-wave P tile (16 q x 32 keys)
  if (threadIdx.x < 32) relS[threadIdx.x] = rel[threadIdx.x];
  __syncthreads();

  const int wv = threadIdx.x >> 5, lane = threadIdx.x & 31;
  const int wave = blockIdx.x * 8 + wv;
  const int b = wave >> 10;                 // 8 heads * 128 qblocks
  const int rem = wave & 1023;
  const int head = rem >> 7;
  const int q0 = (rem & 127) << 4;

  const _Float16* qh = qb + (((size_t)b * CHEADS + head) * CN + q0) * CHD;
  const _Float16* kh = kb + ((size_t)b * CHEADS + head) * CN * CHD;
  const _Float16* vh = vt + ((size_t)b * CHEADS + head) * CHD * CN;

  const v16h aq0 = load_a16(qh + 0,  CHD, lane);
  const v16h aq1 = load_a16(qh + 32, CHD, lane);

  v8f acc[4];
#pragma unroll
  for (int nn = 0; nn < 4; ++nn) acc[nn] = zero8();
  float mrow[8], lrow[8];
#pragma unroll
  for (int r = 0; r < 8; ++r) { mrow[r] = -1e30f; lrow[r] = 0.f; }

  const int col = lane & 15, hi = lane >> 4;
  // query coordinates for the 8 rows this lane-half owns
  int qd[8], qhh[8], qww[8];
#pragma unroll
  for (int r = 0; r < 8; ++r) {
    const int qi = q0 + r + 8 * hi;
    qd[r] = qi >> 8; qhh[r] = (qi >> 4) & 15; qww[r] = qi & 15;
  }
  _Float16* pw = &pS[wv][0];

  for (int kt = 0; kt < CN / 32; ++kt) {
    const int key0 = kt * 32;
    v8f s0 = zero8(), s1 = zero8();
    s0 = wmma16(aq0, load_b16(kh + (size_t)key0 * CHD + 0,  CHD, lane), s0);
    s0 = wmma16(aq1, load_b16(kh + (size_t)key0 * CHD + 32, CHD, lane), s0);
    s1 = wmma16(aq0, load_b16(kh + (size_t)(key0 + 16) * CHD + 0,  CHD, lane), s1);
    s1 = wmma16(aq1, load_b16(kh + (size_t)(key0 + 16) * CHD + 32, CHD, lane), s1);

    const int j0 = key0 + col, j1 = key0 + 16 + col;
    const int j0d = j0 >> 8, j0h = (j0 >> 4) & 15, j0w = j0 & 15;
    const int j1d = j1 >> 8, j1h = (j1 >> 4) & 15, j1w = j1 & 15;

#pragma unroll
    for (int r = 0; r < 8; ++r) {
      // scores = qk/8 + rel_bias
      float dx = (float)(qd[r] - j0d), dy = (float)(qhh[r] - j0h), dz = (float)(qww[r] - j0w);
      int bk0 = (int)(sqrtf(dx * dx + dy * dy + dz * dz) * 0.25f);
      bk0 = bk0 > 31 ? 31 : bk0;
      dx = (float)(qd[r] - j1d); dy = (float)(qhh[r] - j1h); dz = (float)(qww[r] - j1w);
      int bk1 = (int)(sqrtf(dx * dx + dy * dy + dz * dz) * 0.25f);
      bk1 = bk1 > 31 ? 31 : bk1;
      float p0 = s0[r] * 0.125f + relS[bk0];
      float p1 = s1[r] * 0.125f + relS[bk1];

      // online softmax: row max / sum across the 16-lane group
      float mx = fmaxf(p0, p1);
      mx = fmaxf(mx, __shfl_xor(mx, 1, 32));
      mx = fmaxf(mx, __shfl_xor(mx, 2, 32));
      mx = fmaxf(mx, __shfl_xor(mx, 4, 32));
      mx = fmaxf(mx, __shfl_xor(mx, 8, 32));
      const float mnew = fmaxf(mrow[r], mx);
      const float alpha = __expf(mrow[r] - mnew);
      const float e0 = __expf(p0 - mnew), e1 = __expf(p1 - mnew);
      float sum = e0 + e1;
      sum += __shfl_xor(sum, 1, 32);
      sum += __shfl_xor(sum, 2, 32);
      sum += __shfl_xor(sum, 4, 32);
      sum += __shfl_xor(sum, 8, 32);
      lrow[r] = lrow[r] * alpha + sum;
      mrow[r] = mnew;
#pragma unroll
      for (int nn = 0; nn < 4; ++nn) acc[nn][r] *= alpha;

      const int row = r + 8 * hi;
      pw[row * 32 + col]      = (_Float16)e0;
      pw[row * 32 + 16 + col] = (_Float16)e1;
    }

    // acc += P(16x32) x V(32x64); same-wave LDS is in-order, no barrier needed
    const v16h pa = load_a16(pw, 32, lane);
#pragma unroll
    for (int nn = 0; nn < 4; ++nn) {
      v16h bv = load_b16(vh + (size_t)(nn * 16) * CN + key0, CN, lane);
      acc[nn] = wmma16(pa, bv, acc[nn]);
    }
  }

#pragma unroll
  for (int r = 0; r < 8; ++r) {
    const float inv = 1.0f / lrow[r];
    const int qi = q0 + r + 8 * hi;
#pragma unroll
    for (int nn = 0; nn < 4; ++nn) {
      at[((size_t)b * CN + qi) * CC + head * CHD + nn * 16 + col] =
          (_Float16)(acc[nn][r] * inv);
    }
  }
}

// ---------------------------------------------------------------------------
// Projection GEMM + bias + residual:
// out[b][o][n] = sum_c projW[o][c]*attn[b][c][n] + proj_b[o] + x[b][o][n]
// ---------------------------------------------------------------------------
__global__ void proj_gemm(const _Float16* __restrict__ w16, const _Float16* __restrict__ at,
                          const float* __restrict__ pb, const float* __restrict__ x,
                          float* __restrict__ out) {
  const int wave = blockIdx.x * 8 + (threadIdx.x >> 5);
  const int lane = threadIdx.x & 31;
  const int b = wave >> 10;                // 32 Mtiles * 32 Ngroups
  const int rem = wave & 1023;
  const int mt = rem >> 5, nt = rem & 31;
  const _Float16* ab = at + (size_t)b * CN * CC;

  v8f acc[4];
#pragma unroll
  for (int nn = 0; nn < 4; ++nn) acc[nn] = zero8();

  for (int k0 = 0; k0 < CC; k0 += 32) {
    v16h a = load_a16(w16 + (size_t)(mt * 16) * CC + k0, CC, lane);
#pragma unroll
    for (int nn = 0; nn < 4; ++nn) {
      v16h bm = load_b16(ab + (size_t)(nt * 64 + nn * 16) * CC + k0, CC, lane);
      acc[nn] = wmma16(a, bm, acc[nn]);
    }
  }

  const int col = lane & 15, hi = lane >> 4;
#pragma unroll
  for (int nn = 0; nn < 4; ++nn) {
    const int n = nt * 64 + nn * 16 + col;
#pragma unroll
    for (int r = 0; r < 8; ++r) {
      const int o = mt * 16 + r + 8 * hi;
      const size_t idx = ((size_t)b * CC + o) * CN + n;
      out[idx] = acc[nn][r] + pb[o] + x[idx];
    }
  }
}

// ---------------------------------------------------------------------------
// Launch
// ---------------------------------------------------------------------------
extern "C" void kernel_launch(void* const* d_in, const int* in_sizes, int n_in,
                              void* d_out, int out_size, void* d_ws, size_t ws_size,
                              hipStream_t stream) {
  const float* x      = (const float*)d_in[0];
  const float* gn_w   = (const float*)d_in[1];
  const float* gn_b   = (const float*)d_in[2];
  const float* qkv_w  = (const float*)d_in[3];
  const float* qkv_b  = (const float*)d_in[4];
  const float* proj_w = (const float*)d_in[5];
  const float* proj_b = (const float*)d_in[6];
  const float* rel    = (const float*)d_in[7];
  float* out = (float*)d_out;

  char* ws = (char*)d_ws;
  size_t off = 0;
  _Float16* ht   = (_Float16*)(ws + off); off += (size_t)CB * CN * CC * 2;        // 4 MB
  _Float16* qw16 = (_Float16*)(ws + off); off += (size_t)3 * CC * CC * 2;         // 1.5 MB
  _Float16* pw16 = (_Float16*)(ws + off); off += (size_t)CC * CC * 2;             // 0.5 MB
  _Float16* qbuf = (_Float16*)(ws + off); off += (size_t)CB * CHEADS * CN * CHD * 2;
  _Float16* kbuf = (_Float16*)(ws + off); off += (size_t)CB * CHEADS * CN * CHD * 2;
  _Float16* vtb  = (_Float16*)(ws + off); off += (size_t)CB * CHEADS * CHD * CN * 2;
  _Float16* atb  = (_Float16*)(ws + off); off += (size_t)CB * CN * CC * 2;        // total 22 MB

  cvt_f32_f16<<<(3 * CC * CC + 255) / 256, 256, 0, stream>>>(qkv_w, qw16, 3 * CC * CC);
  cvt_f32_f16<<<(CC * CC + 255) / 256, 256, 0, stream>>>(proj_w, pw16, CC * CC);
  gn_kernel<<<16, 256, 0, stream>>>(x, gn_w, gn_b, ht);
  qkv_gemm<<<768, 256, 0, stream>>>(qw16, ht, qkv_b, qbuf, kbuf, vtb);
  attn_kernel<<<256, 256, 0, stream>>>(qbuf, kbuf, vtb, rel, atb);
  proj_gemm<<<256, 256, 0, stream>>>(pw16, atb, proj_b, x, out);
}
